// PointerNetwork_1417339208334
// MI455X (gfx1250) — matrix operand order repaired and checked by reading, not compiled, not run on hardware
//
#include <hip/hip_runtime.h>
#include <hip/hip_bf16.h>
#include <math.h>

// ---------------------------------------------------------------------------
// Pointer network for MI455X (gfx1250, wave32, WMMA).
//   B=64, S=512, IN=64, HID=256, NH=8, HD=32, NL=3
//
// Decoder linearity trick: q = ((dec@Wv+bv)@Wo+bo)@Wq'+bq' = dec@Wf + bf,
// and dec is always dec_start or a row of e, so all candidate score rows are
// precomputed as P_b = (e_b@Wf + bf) @ keys_b^T via WMMA GEMMs. The 512
// sequential steps reduce to masked argmax/softmax over precomputed rows.
// ---------------------------------------------------------------------------

typedef __attribute__((ext_vector_type(16))) __bf16 v16bf;
typedef __attribute__((ext_vector_type(8)))  __bf16 v8bf;
typedef __attribute__((ext_vector_type(4)))  __bf16 v4bf;
typedef __attribute__((ext_vector_type(8)))  float  v8f;    // WMMA C/D
typedef __attribute__((ext_vector_type(8)))  float  v8fl;   // memory vec
typedef __attribute__((ext_vector_type(4)))  float  v4f;

static constexpr int BB   = 64;
static constexpr int SEQ  = 512;
static constexpr int IND  = 64;
static constexpr int HID  = 256;
static constexpr int NH   = 8;
static constexpr int HD   = 32;
static constexpr int BS   = BB * SEQ;   // 32768 rows

__device__ __forceinline__ v8f wmma_bf16(v16bf a, v16bf b, v8f c) {
  // v_wmma_f32_16x16x32_bf16  (neg_a, A, neg_b, B, c_mod, C, reuse_a, reuse_b)
  return __builtin_amdgcn_wmma_f32_16x16x32_bf16(false, a, false, b, (short)0, c,
                                                 false, false);
}

__device__ __forceinline__ v8f zero8() {
  v8f z = {0.f, 0.f, 0.f, 0.f, 0.f, 0.f, 0.f, 0.f};
  return z;
}

__device__ __forceinline__ v16bf cat16(v8bf l, v8bf h) {
  return __builtin_shufflevector(l, h, 0, 1, 2, 3, 4, 5, 6, 7,
                                       8, 9, 10, 11, 12, 13, 14, 15);
}

// A-matrix fragment (16x32, row-major): per ISA 7.12.2 lane-half h holds
// K runs [8h, 8h+8) and [16+8h, 24+8h)  -> two contiguous 16B chunks.
__device__ __forceinline__ v16bf afrag_lds(const __bf16* rowp, int half) {
  const v8bf* p = (const v8bf*)rowp;          // rowp: 32 contiguous bf16
  return cat16(p[half], p[2 + half]);
}
__device__ __forceinline__ v16bf afrag_f32(const float* rowp, int half) {
  v8fl lo = *(const v8fl*)(rowp + half * 8);
  v8fl hi = *(const v8fl*)(rowp + 16 + half * 8);
  return cat16(__builtin_convertvector(lo, v8bf),
               __builtin_convertvector(hi, v8bf));
}
// B-matrix fragment (32x16, KxN): lane = column N, half h holds K runs
// [8h, 8h+8) and [16+8h, 24+8h) (same order as A).
__device__ __forceinline__ v16bf bfrag_lds(const __bf16* colp, int half) {
  const v8bf* p = (const v8bf*)colp;          // colp: 32 contiguous bf16 (K)
  return cat16(p[half], p[2 + half]);
}
__device__ __forceinline__ v16bf bfrag_f32(const float* colp, int half) {
  v8fl lo = *(const v8fl*)(colp + half * 8);
  v8fl hi = *(const v8fl*)(colp + 16 + half * 8);
  return cat16(__builtin_convertvector(lo, v8bf),
               __builtin_convertvector(hi, v8bf));
}

// ---------------------------------------------------------------------------
// Tiled GEMM: out[M,N] = A[M,K](f32) @ W[K,N](f32) [+ bias]  (bf16 WMMA)
// Block 256 thr = 8 waves (4x2), block tile 128x64, wave tile 32x32.
// W tile is stored TRANSPOSED in LDS so B fragments are contiguous b128 loads.
// MODE 0: row-major out. MODE 1: head-split out [B*NH, SEQ, HD].
// ---------------------------------------------------------------------------
template <int MODE, bool HASBIAS>
__global__ __launch_bounds__(256) void gemm_kernel(
    const float* __restrict__ A, const float* __restrict__ W,
    const float* __restrict__ bias, float* __restrict__ out,
    int M, int N, int K) {
  __shared__ __align__(16) __bf16 As[128][32];   // 8 KB, row-major
  __shared__ __align__(16) __bf16 Wst[64][32];   // 4 KB, TRANSPOSED [n][k]

  const int tid  = threadIdx.x;
  const int wave = tid >> 5;
  const int lane = tid & 31;
  const int half = lane >> 4;
  const int idx  = lane & 15;
  const int wm   = wave >> 1;      // 0..3
  const int wn   = wave & 1;       // 0..1
  const int mBlk = blockIdx.x * 128;
  const int nBlk = blockIdx.y * 64;

  v8f acc[2][2];
  acc[0][0] = zero8(); acc[0][1] = zero8();
  acc[1][0] = zero8(); acc[1][1] = zero8();

  for (int k0 = 0; k0 < K; k0 += 32) {
    // Stage A tile: float4 global loads -> bf16x4 LDS stores.
    #pragma unroll
    for (int i = tid * 4; i < 128 * 32; i += 256 * 4) {
      int r = i >> 5, c = i & 31;
      const float* ga = &A[(size_t)(mBlk + r) * K + k0 + c];
      v4f a = *(const v4f*)ga;
      if (k0 + 32 < K) __builtin_prefetch(ga + 32, 0, 1);  // global_prefetch_b8
      *(v4bf*)&As[r][c] = __builtin_convertvector(a, v4bf);
    }
    // Stage W tile transposed: float4 load along N, scatter bf16 along K dim.
    #pragma unroll
    for (int i = tid * 4; i < 32 * 64; i += 256 * 4) {
      int r = i >> 6, c = i & 63;
      const float* gw = &W[(size_t)(k0 + r) * N + nBlk + c];
      v4f w = *(const v4f*)gw;
      if (k0 + 32 < K) __builtin_prefetch(gw + (size_t)32 * N, 0, 1);
      v4bf wb = __builtin_convertvector(w, v4bf);
      #pragma unroll
      for (int u = 0; u < 4; u++) Wst[c + u][r] = wb[u];
    }
    __syncthreads();

    #pragma unroll
    for (int tm = 0; tm < 2; tm++) {
      v16bf af = afrag_lds(&As[wm * 32 + tm * 16 + idx][0], half);
      #pragma unroll
      for (int tn = 0; tn < 2; tn++) {
        v16bf bf = bfrag_lds(&Wst[wn * 32 + tn * 16 + idx][0], half);
        acc[tm][tn] = wmma_bf16(af, bf, acc[tm][tn]);
      }
    }
    __syncthreads();
  }

  // Epilogue: C/D layout — VGPR r, lanes0-15 -> M=r, lanes16-31 -> M=8+r.
  #pragma unroll
  for (int tn = 0; tn < 2; tn++) {
    #pragma unroll
    for (int tm = 0; tm < 2; tm++) {
      int n = nBlk + wn * 32 + tn * 16 + idx;
      float bn = HASBIAS ? bias[n] : 0.f;
      #pragma unroll
      for (int r = 0; r < 8; r++) {
        int m = mBlk + wm * 32 + tm * 16 + half * 8 + r;
        float v = acc[tm][tn][r] + bn;
        if (MODE == 0) {
          out[(size_t)m * N + n] = v;
        } else {  // head-split: [B*NH, SEQ, HD]
          int b = m >> 9, s = m & (SEQ - 1);
          int h = n >> 5, hd = n & (HD - 1);
          out[(((size_t)(b * NH + h)) * SEQ + s) * HD + hd] = v;
        }
      }
    }
  }
}

// ---------------------------------------------------------------------------
// Fused encoder attention for one (b,h) and 32 query rows.
// Block 64 threads = 2 waves; each wave owns 16 query rows.
// scores (WMMA, K=HD=32) -> softmax in LDS -> ctx = P@V (WMMA).
// ---------------------------------------------------------------------------
__global__ __launch_bounds__(64) void attn_kernel(
    const float* __restrict__ Q, const float* __restrict__ K,
    const float* __restrict__ V, float* __restrict__ ctx) {
  __shared__ __align__(32) float sc[32][SEQ];   // 64 KB score/prob buffer

  const int bh    = blockIdx.x;            // b*NH + h
  const int qBase = blockIdx.y * 32;
  const int wave  = threadIdx.x >> 5;
  const int lane  = threadIdx.x & 31;
  const int half  = lane >> 4, idx = lane & 15;
  const int rowLoc = wave * 16;
  const float scale = 0.17677669529663687f;  // 1/sqrt(32)

  const float* Qb = Q + (size_t)bh * SEQ * HD;
  const float* Kb = K + (size_t)bh * SEQ * HD;
  const float* Vb = V + (size_t)bh * SEQ * HD;

  // A fragment: this wave's 16 query rows (fixed across key tiles).
  v16bf aq = afrag_f32(Qb + (size_t)(qBase + rowLoc + idx) * HD, half);

  // Phase 1: scores = Q @ K^T / sqrt(HD)
  for (int nt = 0; nt < SEQ / 16; nt++) {
    v16bf bk = bfrag_f32(Kb + (size_t)(nt * 16 + idx) * HD, half);
    v8f s = zero8();
    s = wmma_bf16(aq, bk, s);
    #pragma unroll
    for (int r = 0; r < 8; r++)
      sc[rowLoc + half * 8 + r][nt * 16 + idx] = s[r] * scale;
  }
  __syncthreads();

  // Phase 2: row softmax (wave32 shuffle reductions), probs back into sc.
  for (int r = 0; r < 16; r++) {
    float* srow = sc[rowLoc + r];
    float mx = -1e30f;
    #pragma unroll
    for (int i = 0; i < 16; i++) mx = fmaxf(mx, srow[lane + i * 32]);
    for (int m = 16; m > 0; m >>= 1) mx = fmaxf(mx, __shfl_xor(mx, m, 32));
    float sum = 0.f;
    #pragma unroll
    for (int i = 0; i < 16; i++) {
      float e = __expf(srow[lane + i * 32] - mx);
      srow[lane + i * 32] = e;
      sum += e;
    }
    for (int m = 16; m > 0; m >>= 1) sum += __shfl_xor(sum, m, 32);
    float inv = 1.f / sum;
    #pragma unroll
    for (int i = 0; i < 16; i++) srow[lane + i * 32] *= inv;
  }
  __syncthreads();

  // Phase 3: ctx = P @ V  (N = HD = 32 -> two 16-wide column tiles)
  v8f acc0 = zero8(), acc1 = zero8();
  for (int kt = 0; kt < SEQ / 32; kt++) {
    v16bf ap = afrag_f32(&sc[rowLoc + idx][kt * 32], half);
    v16bf b0, b1;
    #pragma unroll
    for (int j = 0; j < 16; j++) {          // V gather is strided (row varies)
      int key = kt * 32 + (j < 8 ? half * 8 + j : 16 + half * 8 + (j - 8));
      b0[j] = (__bf16)Vb[(size_t)key * HD + idx];
      b1[j] = (__bf16)Vb[(size_t)key * HD + 16 + idx];
    }
    acc0 = wmma_bf16(ap, b0, acc0);
    acc1 = wmma_bf16(ap, b1, acc1);
  }

  // Store ctx in [B, S, HID] layout (heads re-merged).
  int b = bh >> 3, h = bh & 7;
  #pragma unroll
  for (int r = 0; r < 8; r++) {
    int s = qBase + rowLoc + half * 8 + r;
    float* dst = ctx + ((size_t)(b * SEQ + s)) * HID + h * HD;
    dst[idx]      = acc0[r];
    dst[16 + idx] = acc1[r];
  }
}

// ---------------------------------------------------------------------------
// Residual + LayerNorm: e = LN(add + e) * g + b.   One wave per 256-wide row.
// ---------------------------------------------------------------------------
__global__ __launch_bounds__(128) void ln_kernel(
    const float* __restrict__ add, float* __restrict__ e,
    const float* __restrict__ g, const float* __restrict__ bv, int rows) {
  int wave = threadIdx.x >> 5, lane = threadIdx.x & 31;
  int row  = blockIdx.x * 4 + wave;
  if (row >= rows) return;
  const float* a = add + (size_t)row * HID;
  float* er = e + (size_t)row * HID;

  float x[8];
  float sum = 0.f;
  #pragma unroll
  for (int i = 0; i < 8; i++) {
    x[i] = a[lane + i * 32] + er[lane + i * 32];
    sum += x[i];
  }
  for (int m = 16; m > 0; m >>= 1) sum += __shfl_xor(sum, m, 32);
  float mean = sum * (1.f / HID);
  float var = 0.f;
  #pragma unroll
  for (int i = 0; i < 8; i++) { float d = x[i] - mean; var += d * d; }
  for (int m = 16; m > 0; m >>= 1) var += __shfl_xor(var, m, 32);
  float rs = rsqrtf(var * (1.f / HID) + 1e-5f);
  #pragma unroll
  for (int i = 0; i < 8; i++) {
    int c = lane + i * 32;
    er[c] = (x[i] - mean) * rs * g[c] + bv[c];
  }
}

// ---------------------------------------------------------------------------
// P_b = Qall_b[512,256] @ keys_b^T[256,512]   (batched WMMA GEMM, direct
// global fragments — everything is L2-resident on a 192MB L2).
// Grid (4, 8, B); block 256 thr, tile 128x64.
// ---------------------------------------------------------------------------
__global__ __launch_bounds__(256) void pscore_kernel(
    const float* __restrict__ Qall, const float* __restrict__ keys,
    float* __restrict__ P) {
  const int b = blockIdx.z;
  const int tid = threadIdx.x, wave = tid >> 5, lane = tid & 31;
  const int half = lane >> 4, idx = lane & 15;
  const int wm = wave >> 1, wn = wave & 1;
  const int mBlk = blockIdx.x * 128, nBlk = blockIdx.y * 64;

  const float* Qb = Qall + (size_t)b * SEQ * HID;
  const float* Kb = keys + (size_t)b * SEQ * HID;
  float* Pb = P + (size_t)b * SEQ * SEQ;

  v8f acc[2][2];
  acc[0][0] = zero8(); acc[0][1] = zero8();
  acc[1][0] = zero8(); acc[1][1] = zero8();

  for (int k0 = 0; k0 < HID; k0 += 32) {
    v16bf af[2], bf[2];
    #pragma unroll
    for (int tm = 0; tm < 2; tm++)
      af[tm] = afrag_f32(Qb + (size_t)(mBlk + wm * 32 + tm * 16 + idx) * HID + k0, half);
    #pragma unroll
    for (int tn = 0; tn < 2; tn++)   // B[k][n] = keys[n][k]
      bf[tn] = bfrag_f32(Kb + (size_t)(nBlk + wn * 32 + tn * 16 + idx) * HID + k0, half);
    #pragma unroll
    for (int tm = 0; tm < 2; tm++)
      #pragma unroll
      for (int tn = 0; tn < 2; tn++)
        acc[tm][tn] = wmma_bf16(af[tm], bf[tn], acc[tm][tn]);
  }
  #pragma unroll
  for (int tm = 0; tm < 2; tm++)
    #pragma unroll
    for (int tn = 0; tn < 2; tn++)
      #pragma unroll
      for (int r = 0; r < 8; r++) {
        int m = mBlk + wm * 32 + tm * 16 + half * 8 + r;
        int n = nBlk + wn * 32 + tn * 16 + idx;
        Pb[(size_t)m * SEQ + n] = acc[tm][tn][r];
      }
}

// ---------------------------------------------------------------------------
// Small helpers
// ---------------------------------------------------------------------------
__global__ void matvec_kernel(const float* __restrict__ v,
                              const float* __restrict__ W,
                              const float* __restrict__ bias,
                              float* __restrict__ out, int K, int N) {
  int n = blockIdx.x * blockDim.x + threadIdx.x;
  if (n >= N) return;
  float s = bias ? bias[n] : 0.f;
  for (int k = 0; k < K; k++) s += v[k] * W[(size_t)k * N + n];
  out[n] = s;
}

__global__ void s0_kernel(const float* __restrict__ q0,
                          const float* __restrict__ keys,
                          float* __restrict__ s0, int total) {
  int i = blockIdx.x * 256 + threadIdx.x;
  if (i >= total) return;
  const float* kr = keys + (size_t)i * HID;
  float s = 0.f;
  for (int k = 0; k < HID; k++) s += q0[k] * kr[k];
  s0[i] = s;
}

// ---------------------------------------------------------------------------
// Sequential pointer decoder: one block per batch, 512 dependent steps over
// precomputed score rows. Wave32 shuffle reductions + tiny cross-wave LDS
// stage -> only 4 barriers per step.
// ---------------------------------------------------------------------------
__global__ __launch_bounds__(256) void decoder_kernel(
    const float* __restrict__ P, const float* __restrict__ s0,
    int* __restrict__ outIdx, float* __restrict__ outLogp) {
  __shared__ unsigned maskBits[16];
  __shared__ float wV[8];
  __shared__ int   wI[8];
  __shared__ float wS[8];
  __shared__ float bcV;
  __shared__ int   bcI;

  const int b = blockIdx.x, tid = threadIdx.x;
  const int lane = tid & 31, wave = tid >> 5;
  if (tid < 16) maskBits[tid] = 0xFFFFFFFFu;
  __syncthreads();

  const float* row = s0 + (size_t)b * SEQ;
  float logpAcc = 0.f;  // only thread 0's value is used

  for (int t = 0; t < SEQ; t++) {
    int j0 = tid, j1 = tid + 256;
    float v0 = row[j0], v1 = row[j1];
    bool m0 = (maskBits[j0 >> 5] >> (j0 & 31)) & 1u;
    bool m1 = (maskBits[j1 >> 5] >> (j1 & 31)) & 1u;
    float a0 = m0 ? v0 : -1e9f;
    float a1 = m1 ? v1 : -1e9f;

    // wave-level masked argmax (ties -> lowest index, matches jnp.argmax)
    float bv; int bi;
    if (a1 > a0) { bv = a1; bi = j1; } else { bv = a0; bi = j0; }
    for (int m = 16; m > 0; m >>= 1) {
      float ov = __shfl_xor(bv, m, 32);
      int   oi = __shfl_xor(bi, m, 32);
      if (ov > bv || (ov == bv && oi < bi)) { bv = ov; bi = oi; }
    }
    if (lane == 0) { wV[wave] = bv; wI[wave] = bi; }
    __syncthreads();
    if (tid == 0) {
      float mv = wV[0]; int mi = wI[0];
      #pragma unroll
      for (int w = 1; w < 8; w++) {
        if (wV[w] > mv || (wV[w] == mv && wI[w] < mi)) { mv = wV[w]; mi = wI[w]; }
      }
      bcV = mv; bcI = mi;
    }
    __syncthreads();
    float mx = bcV;
    int idx  = bcI;

    // denom = sum exp(masked - max); masked terms underflow to 0 (matches ref)
    float es = __expf(a0 - mx) + __expf(a1 - mx);
    for (int m = 16; m > 0; m >>= 1) es += __shfl_xor(es, m, 32);
    if (lane == 0) wS[wave] = es;
    __syncthreads();
    if (tid == 0) {
      float denom = 0.f;
      #pragma unroll
      for (int w = 0; w < 8; w++) denom += wS[w];
      logpAcc += __logf(1.f / denom + 1e-10f);  // p[idx] = 1/denom
      outIdx[(size_t)b * SEQ + t] = idx;
      maskBits[idx >> 5] &= ~(1u << (idx & 31));
    }
    __syncthreads();
    row = P + ((size_t)(b * SEQ + idx)) * SEQ;  // dec = e[b, idx]
  }
  if (tid == 0) outLogp[b] = logpAcc;
}

// ---------------------------------------------------------------------------
// Host orchestration.
// Input order = JAX pytree (sorted dict keys) flattening of setup_inputs():
//  0:x | dec_attn{ 1:k.b 2:k.w 3:o.b 4:o.w 5:q.b 6:q.w 7:v.b 8:v.w } |
//  9:dec_start | 10:embed.b 11:embed.w |
//  enc[l] base 12+8l: {k.b,k.w,o.b,o.w,q.b,q.w,v.b,v.w} |
//  36:ln_b 37:ln_g | 38:ptr_k.b 39:ptr_k.w | 40:ptr_q.b 41:ptr_q.w
// ---------------------------------------------------------------------------
extern "C" void kernel_launch(void* const* d_in, const int* in_sizes, int n_in,
                              void* d_out, int out_size, void* d_ws, size_t ws_size,
                              hipStream_t stream) {
  (void)in_sizes; (void)n_in; (void)out_size; (void)ws_size;
  auto F = [&](int i) { return (const float*)d_in[i]; };
  const float* x = F(0);

  // Workspace layout (floats).
  float* ws   = (float*)d_ws;
  const size_t RC = (size_t)BS * HID;  // 8,388,608 floats per activation buf
  float* e    = ws;             // encoder state
  float* bufQ = e    + RC;      // Q  -> later keys
  float* bufK = bufQ + RC;      // K  -> later Qall
  float* bufV = bufK + RC;      // V  -> later start of P (needs 2*RC)
  float* ctxb = bufV + RC;      // attn ctx (merged into P region later)
  float* tmpo = ctxb + RC;      // o-projection output
  float* P    = bufV;           // [B, S, S] = exactly 2*RC floats
  float* Wtmp = tmpo + RC;      // 256x256
  float* Wf   = Wtmp + HID * HID;
  float* bv2  = Wf   + HID * HID;
  float* bfu  = bv2  + HID;
  float* q0   = bfu  + HID;
  float* s0v  = q0   + HID;     // [B*S]

  dim3 blk256(256), blk128(128), blk64(64);
  dim3 gBig(BS / 128, HID / 64);       // (256, 4) — activation GEMMs
  dim3 gWgt(HID / 128, HID / 64);      // (2, 4)   — weight-weight GEMMs

  // Encoder: embed
  gemm_kernel<0, true><<<gBig, blk256, 0, stream>>>(x, F(11), F(10), e, BS, HID, IND);

  // Encoder: 3 post-norm self-attention blocks
  for (int l = 0; l < 3; l++) {
    int p = 12 + l * 8;  // k.b,k.w,o.b,o.w,q.b,q.w,v.b,v.w
    gemm_kernel<1, true><<<gBig, blk256, 0, stream>>>(e, F(p + 5), F(p + 4), bufQ, BS, HID, HID);
    gemm_kernel<1, true><<<gBig, blk256, 0, stream>>>(e, F(p + 1), F(p + 0), bufK, BS, HID, HID);
    gemm_kernel<1, true><<<gBig, blk256, 0, stream>>>(e, F(p + 7), F(p + 6), bufV, BS, HID, HID);
    attn_kernel<<<dim3(BB * NH, SEQ / 32), blk64, 0, stream>>>(bufQ, bufK, bufV, ctxb);
    gemm_kernel<0, true><<<gBig, blk256, 0, stream>>>(ctxb, F(p + 3), F(p + 2), tmpo, BS, HID, HID);
    ln_kernel<<<BS / 4, blk128, 0, stream>>>(tmpo, e, F(37), F(36), BS);
  }

  // keys = e @ ptr_k.w + ptr_k.b   ([B,S,HID] row-major)
  gemm_kernel<0, true><<<gBig, blk256, 0, stream>>>(e, F(39), F(38), bufQ, BS, HID, HID);

  // Fused decoder weights: Wf = Wv @ Wo @ Wptr_q ; bfu = (bv@Wo+bo)@Wptr_q + bptr_q
  gemm_kernel<0, false><<<gWgt, blk256, 0, stream>>>(F(8), F(4), nullptr, Wtmp, HID, HID, HID);
  gemm_kernel<0, false><<<gWgt, blk256, 0, stream>>>(Wtmp, F(41), nullptr, Wf, HID, HID, HID);
  matvec_kernel<<<1, blk256, 0, stream>>>(F(7), F(4), F(3), bv2, HID, HID);
  matvec_kernel<<<1, blk256, 0, stream>>>(bv2, F(41), F(40), bfu, HID, HID);
  matvec_kernel<<<1, blk256, 0, stream>>>(F(9), Wf, bfu, q0, HID, HID);  // dec_start row

  // Qall = e @ Wf + bfu ; P_b = Qall_b @ keys_b^T ; s0_b = q0 . keys_b
  gemm_kernel<0, true><<<gBig, blk256, 0, stream>>>(e, Wf, bfu, bufK, BS, HID, HID);
  pscore_kernel<<<dim3(SEQ / 128, SEQ / 64, BB), blk256, 0, stream>>>(bufK, bufQ, P);
  s0_kernel<<<BS / 256, blk256, 0, stream>>>(q0, bufQ, s0v, BS);

  // Sequential pointer decoding: outputs idx [B,S] (int) then logp-sums [B].
  decoder_kernel<<<BB, blk256, 0, stream>>>(P, s0v, (int*)d_out,
                                            (float*)d_out + (size_t)BB * SEQ);
}
